// Attention_16827681866234
// MI455X (gfx1250) — compile-verified, matrix operand order checked by reading, not run on hardware
//
#include <hip/hip_runtime.h>
#include <hip/hip_bf16.h>

#define BB 2
#define LL 2048
#define HH 1024
#define NHEAD 16
#define DD 64
#define MROWS (BB*LL)        // 4096
#define LOG2E 1.44269504088896340736f

typedef _Float16 f16;
typedef __attribute__((ext_vector_type(16))) _Float16 v16h;
typedef __attribute__((ext_vector_type(8)))  _Float16 v8h;
typedef __attribute__((ext_vector_type(8)))  float    v8f;
typedef __attribute__((address_space(3)))    f16      lds_f16;

// ---------------------------------------------------------------------------
// LDS byte offset (wave-relative) of a generic pointer to __shared__ memory.
// ---------------------------------------------------------------------------
__device__ __forceinline__ unsigned int lds_off(const void* p) {
    return (unsigned int)(unsigned long long)(lds_f16*)(void*)(p);
}

// Async global->LDS copy, 16 bytes per lane (gfx1250, tracked by ASYNCcnt).
__device__ __forceinline__ void async_load_b128(unsigned int lds_byte_off,
                                                const void* gaddr) {
    asm volatile("global_load_async_to_lds_b128 %0, %1, off"
                 :: "v"(lds_byte_off), "v"((unsigned long long)(uintptr_t)gaddr)
                 : "memory");
}
__device__ __forceinline__ void wait_async0() {
    asm volatile("s_wait_asynccnt 0x0" ::: "memory");
}
__device__ __forceinline__ void wait_async4() {   // retire oldest 4 (prev chunk)
    asm volatile("s_wait_asynccnt 0x4" ::: "memory");
}

// ---------------------------------------------------------------------------
// Fused DPP row-rotate max-reduction over the 16-lane DPP row: each step is a
// single v_max_num_f32 with DPP on src0 (VOP2+DPP16). Pure VALU, no waits.
// ---------------------------------------------------------------------------
__device__ __forceinline__ float rowmax16(float x) {
    asm("v_max_num_f32_dpp %0, %0, %0 row_ror:1 row_mask:0xf bank_mask:0xf bound_ctrl:1\n\t"
        "v_max_num_f32_dpp %0, %0, %0 row_ror:2 row_mask:0xf bank_mask:0xf bound_ctrl:1\n\t"
        "v_max_num_f32_dpp %0, %0, %0 row_ror:4 row_mask:0xf bank_mask:0xf bound_ctrl:1\n\t"
        "v_max_num_f32_dpp %0, %0, %0 row_ror:8 row_mask:0xf bank_mask:0xf bound_ctrl:1"
        : "+v"(x));
    return x;
}

// ---------------------------------------------------------------------------
// Fragment loader: both A (16 rows x K32) and B (K32 x 16 cols, stored as
// [col][k] row-major) use the same per-lane layout on gfx1250 wave32:
//   lr = l&15, hf = l>>4
//   elements 0..7  : K = hf*8 + {0..7};  elements 8..15 : K = 16+hf*8+{0..7}
// ---------------------------------------------------------------------------
__device__ __forceinline__ v16h load_frag(const f16* base, int stride) {
    const int lane = threadIdx.x & 31;
    const int lr = lane & 15, hf = lane >> 4;
    const f16* p = base + lr * stride + hf * 8;
    v8h lo = *(const v8h*)(p);
    v8h hi = *(const v8h*)(p + 16);
    return __builtin_shufflevector(lo, hi, 0,1,2,3,4,5,6,7,8,9,10,11,12,13,14,15);
}

// ---------------------------------------------------------------------------
// f32 -> f16 elementwise convert
// ---------------------------------------------------------------------------
__global__ void cvt_f16_kernel(const float* __restrict__ in,
                               unsigned short* __restrict__ out_, int n) {
    f16* out = (f16*)out_;
    int i = blockIdx.x * blockDim.x + threadIdx.x;
    if (i < n) out[i] = (f16)in[i];
}

// ---------------------------------------------------------------------------
// 1024x1024 transpose + convert: out[c][r] = (f16) in[r][c]
// ---------------------------------------------------------------------------
__global__ __launch_bounds__(256)
void cvt_transpose_kernel(const float* __restrict__ in,
                          unsigned short* __restrict__ out_) {
    f16* out = (f16*)out_;
    __shared__ float tile[32][33];
    const int bx = blockIdx.x * 32, by = blockIdx.y * 32;
    const int tx = threadIdx.x & 31, ty = threadIdx.x >> 5;   // ty 0..7
#pragma unroll
    for (int r = ty; r < 32; r += 8)
        tile[r][tx] = in[(by + r) * HH + bx + tx];
    __syncthreads();
#pragma unroll
    for (int r = ty; r < 32; r += 8)
        out[(bx + r) * HH + by + tx] = (f16)tile[tx][r];
}

// ---------------------------------------------------------------------------
// Generic f16 WMMA GEMM:  C[M=4096, 1024] = A[M,1024] * BT[1024,1024]^T
//   A  : [M][K]    row-major f16;  BT : [col][K] row-major f16
// Block tile 128x128, 8 waves (4 in M, 2 in N), wave tile 32x64, K-step 32.
// Double-buffered async global->LDS staging (ASYNCcnt pipelining).
// MODE 0: f16 out, head-major  q[((b*NH+n)*L + f)*D + d]   (Q/K), *scale
// MODE 1: f16 out, V transposed v[((b*NH+n)*D + d)*L + t]
// MODE 2: f32 out, plain row-major (final output)
// ---------------------------------------------------------------------------
#define GLDA 40   // padded LDS row stride (halfs): 80B, 16B-aligned

__device__ __forceinline__ void stage_gemm(const f16* gA, const f16* gB, int k0,
                                           unsigned int oA, unsigned int oB) {
    async_load_b128(oA,      gA + k0);
    async_load_b128(oA + 16, gA + k0 + 8);
    async_load_b128(oB,      gB + k0);
    async_load_b128(oB + 16, gB + k0 + 8);
}

template<int MODE>
__global__ __launch_bounds__(256)
void gemm_f16_kernel(const unsigned short* __restrict__ A_,
                     const unsigned short* __restrict__ BT_,
                     void* __restrict__ Cout, float scale) {
    const f16* A  = (const f16*)A_;
    const f16* BT = (const f16*)BT_;
    __shared__ f16 sA[2][128 * GLDA];
    __shared__ f16 sB[2][128 * GLDA];

    const int tid  = threadIdx.x;
    const int wid  = tid >> 5;
    const int lane = tid & 31;
    const int lr = lane & 15, hf = lane >> 4;
    const int bm = blockIdx.y * 128;
    const int bn = blockIdx.x * 128;
    const int wm = (wid & 3) * 32;
    const int wn = (wid >> 2) * 64;

    v8f acc[2][4];
#pragma unroll
    for (int i = 0; i < 2; i++)
#pragma unroll
        for (int j = 0; j < 4; j++) acc[i][j] = (v8f)(0.0f);

    const int rowA = tid >> 1;            // 0..127
    const int segA = (tid & 1) * 16;      // 0 or 16
    const f16* gA = A  + (size_t)(bm + rowA) * HH + segA;
    const f16* gB = BT + (size_t)(bn + rowA) * HH + segA;
    const unsigned int oA[2] = { lds_off(&sA[0][rowA * GLDA + segA]),
                                 lds_off(&sA[1][rowA * GLDA + segA]) };
    const unsigned int oB[2] = { lds_off(&sB[0][rowA * GLDA + segA]),
                                 lds_off(&sB[1][rowA * GLDA + segA]) };

    stage_gemm(gA, gB, 0, oA[0], oB[0]);        // prologue: chunk 0 in flight

    for (int k0 = 0; k0 < HH; k0 += 32) {
        const int cur = (k0 >> 5) & 1;
        const bool more = (k0 + 32 < HH);
        if (more) stage_gemm(gA, gB, k0 + 32, oA[cur ^ 1], oB[cur ^ 1]);
        if (more) wait_async4(); else wait_async0();
        __syncthreads();

        v16h af[2], bf[4];
#pragma unroll
        for (int i = 0; i < 2; i++) af[i] = load_frag(&sA[cur][(wm + i * 16) * GLDA], GLDA);
#pragma unroll
        for (int j = 0; j < 4; j++) bf[j] = load_frag(&sB[cur][(wn + j * 16) * GLDA], GLDA);
#pragma unroll
        for (int i = 0; i < 2; i++)
#pragma unroll
            for (int j = 0; j < 4; j++)
                acc[i][j] = __builtin_amdgcn_wmma_f32_16x16x32_f16(
                    false, af[i], false, bf[j], (short)0, acc[i][j], false, false);
        __syncthreads();   // all reads of sA/sB[cur] done before it is re-staged
    }

    // Epilogue (compile-time MODE -> straight-line stores)
#pragma unroll
    for (int i = 0; i < 2; i++) {
#pragma unroll
        for (int j = 0; j < 4; j++) {
#pragma unroll
            for (int vg = 0; vg < 8; vg++) {
                const int row = bm + wm + i * 16 + vg + 8 * hf;
                const int col = bn + wn + j * 16 + lr;
                const float v = acc[i][j][vg] * scale;
                if (MODE == 2) {
                    ((float*)Cout)[(size_t)row * HH + col] = v;
                } else {
                    const int b = row >> 11, f = row & (LL - 1);
                    const int n = col >> 6,  d = col & (DD - 1);
                    f16* o = (f16*)Cout;
                    if (MODE == 0)
                        o[(((size_t)(b * NHEAD + n)) * LL + f) * DD + d] = (f16)v;
                    else
                        o[(((size_t)(b * NHEAD + n)) * DD + d) * LL + f] = (f16)v;
                }
            }
        }
    }
}

// ---------------------------------------------------------------------------
// Flash attention: block = (f-tile of 128, head, batch); 8 waves x 16 rows.
// Q head-major [bh][L][D] (pre-scaled by log2e/sqrt(D)), K head-major,
// V transposed [bh][D][L]. Double-buffered async K/V staging. Per 64-wide T
// chunk: S = Q*K^T (8 WMMA) + bias*log2e (fma), online softmax entirely in
// the log2 domain (v_exp_f32 is base-2: no extra muls), row-max via fused
// DPP v_max reductions; denominator via ones-column WMMA; P via LDS;
// O += P*V (8 WMMA) + 2 ones-WMMA.
// ---------------------------------------------------------------------------
#define PST 72   // padded LDS stride (halfs): 144B, 16B-aligned

__device__ __forceinline__ void stage_attn(const f16* Kbh, const f16* Vbh, int t0,
                                           int ldr, int seg,
                                           unsigned int oK, unsigned int oV) {
    const f16* gK = Kbh + (size_t)(t0 + ldr) * DD + seg;
    const f16* gV = Vbh + (size_t)ldr * LL + t0 + seg;
    async_load_b128(oK,      gK);
    async_load_b128(oK + 16, gK + 8);
    async_load_b128(oV,      gV);
    async_load_b128(oV + 16, gV + 8);
}

__global__ __launch_bounds__(256)
void attn_kernel(const unsigned short* __restrict__ Q_,
                 const unsigned short* __restrict__ K_,
                 const unsigned short* __restrict__ VT_,
                 const float* __restrict__ bias,
                 unsigned short* __restrict__ attnOut_) {
    const f16* Q  = (const f16*)Q_;
    const f16* K  = (const f16*)K_;
    const f16* VT = (const f16*)VT_;
    f16* attnOut  = (f16*)attnOut_;

    __shared__ f16 sK[2][64 * PST];       // [t][d]
    __shared__ f16 sV[2][64 * PST];       // [d][t]
    __shared__ f16 sP[8 * 16 * PST];      // per-wave [16][PST]

    const int tid = threadIdx.x, wid = tid >> 5, lane = tid & 31;
    const int lr = lane & 15, hf = lane >> 4;
    const int ft = blockIdx.x, head = blockIdx.y, b = blockIdx.z;
    const int bh = b * NHEAD + head;

    const f16* Qbh = Q  + (size_t)bh * LL * DD;
    const f16* Kbh = K  + (size_t)bh * LL * DD;
    const f16* Vbh = VT + (size_t)bh * DD * LL;
    const float* biasb = bias + (size_t)b * LL * LL;

    const int f0 = ft * 128 + wid * 16;

    // Q fragments for this wave's 16 rows (loaded once from global)
    v16h qf[2];
#pragma unroll
    for (int kc = 0; kc < 2; kc++)
        qf[kc] = load_frag(Qbh + (size_t)f0 * DD + kc * 32, DD);

    // constant all-ones B fragment: row-sum accumulator via WMMA
    v16h ones;
#pragma unroll
    for (int i = 0; i < 16; i++) ones[i] = (f16)1.0f;

    v8f O[4], O5;
#pragma unroll
    for (int dc = 0; dc < 4; dc++) O[dc] = (v8f)(0.0f);
    O5 = (v8f)(0.0f);
    float mrun[8];
#pragma unroll
    for (int vg = 0; vg < 8; vg++) mrun[vg] = -1e30f;

    const int ldr = tid >> 2;             // 0..63
    const int seg = (tid & 3) * 16;       // 0,16,32,48
    f16* myP = sP + wid * 16 * PST;

    const unsigned int oK[2] = { lds_off(&sK[0][ldr * PST + seg]),
                                 lds_off(&sK[1][ldr * PST + seg]) };
    const unsigned int oV[2] = { lds_off(&sV[0][ldr * PST + seg]),
                                 lds_off(&sV[1][ldr * PST + seg]) };

    stage_attn(Kbh, Vbh, 0, ldr, seg, oK[0], oV[0]);   // chunk 0 in flight

    for (int t0 = 0; t0 < LL; t0 += 64) {
        const int cur = (t0 >> 6) & 1;
        const bool more = (t0 + 64 < LL);
        if (more) stage_attn(Kbh, Vbh, t0 + 64, ldr, seg, oK[cur ^ 1], oV[cur ^ 1]);
        if (more) {   // L2 warm-up two chunks ahead
            __builtin_prefetch(Kbh + (size_t)(t0 + 128 + ldr) * DD + seg, 0, 0);
            __builtin_prefetch(Vbh + (size_t)ldr * LL + t0 + 128 + seg, 0, 0);
        }
        if (more) wait_async4(); else wait_async0();
        __syncthreads();

        // S = Q * K^T : preload all 8 B fragments, then 8 WMMAs
        v16h kb[4][2];
#pragma unroll
        for (int tc = 0; tc < 4; tc++)
#pragma unroll
            for (int kc = 0; kc < 2; kc++)
                kb[tc][kc] = load_frag(&sK[cur][(tc * 16) * PST + kc * 32], PST);

        v8f S[4];
#pragma unroll
        for (int tc = 0; tc < 4; tc++) S[tc] = (v8f)(0.0f);
#pragma unroll
        for (int kc = 0; kc < 2; kc++)
#pragma unroll
            for (int tc = 0; tc < 4; tc++)
                S[tc] = __builtin_amdgcn_wmma_f32_16x16x32_f16(
                    false, qf[kc], false, kb[tc][kc], (short)0, S[tc], false, false);

        // + bias*log2e (S is in the log2 domain; single fma per element)
#pragma unroll
        for (int tc = 0; tc < 4; tc++)
#pragma unroll
            for (int vg = 0; vg < 8; vg++)
                S[tc][vg] = fmaf(
                    biasb[(size_t)(f0 + vg + 8 * hf) * LL + t0 + tc * 16 + lr],
                    LOG2E, S[tc][vg]);

        // row max: local max over 4 col-tiles, then fused DPP reduction
        float sc[8];
#pragma unroll
        for (int vg = 0; vg < 8; vg++) {
            float mv = fmaxf(fmaxf(S[0][vg], S[1][vg]), fmaxf(S[2][vg], S[3][vg]));
            mv = rowmax16(mv);
            const float mnew = fmaxf(mrun[vg], mv);
            sc[vg] = __builtin_amdgcn_exp2f(mrun[vg] - mnew);
            mrun[vg] = mnew;
        }

        // P = 2^(S - m); rescale accumulators (incl. denominator accum)
#pragma unroll
        for (int tc = 0; tc < 4; tc++)
#pragma unroll
            for (int vg = 0; vg < 8; vg++)
                S[tc][vg] = __builtin_amdgcn_exp2f(S[tc][vg] - mrun[vg]);
#pragma unroll
        for (int dc = 0; dc < 4; dc++)
#pragma unroll
            for (int vg = 0; vg < 8; vg++)
                O[dc][vg] *= sc[vg];
#pragma unroll
        for (int vg = 0; vg < 8; vg++) O5[vg] *= sc[vg];

        // stage P (C-layout -> A-layout via LDS)
#pragma unroll
        for (int tc = 0; tc < 4; tc++)
#pragma unroll
            for (int vg = 0; vg < 8; vg++)
                myP[(vg + 8 * hf) * PST + tc * 16 + lr] = (f16)S[tc][vg];

        // O += P * V ; O5 += P * ones  (denominator, no shuffle reduction)
        v16h pa[2];
#pragma unroll
        for (int tk = 0; tk < 2; tk++) pa[tk] = load_frag(myP + tk * 32, PST);
        v16h vb[4][2];
#pragma unroll
        for (int dc = 0; dc < 4; dc++)
#pragma unroll
            for (int tk = 0; tk < 2; tk++)
                vb[dc][tk] = load_frag(&sV[cur][(dc * 16) * PST + tk * 32], PST);
#pragma unroll
        for (int tk = 0; tk < 2; tk++) {
#pragma unroll
            for (int dc = 0; dc < 4; dc++)
                O[dc] = __builtin_amdgcn_wmma_f32_16x16x32_f16(
                    false, pa[tk], false, vb[dc][tk], (short)0, O[dc], false, false);
            O5 = __builtin_amdgcn_wmma_f32_16x16x32_f16(
                false, pa[tk], false, ones, (short)0, O5, false, false);
        }
        __syncthreads();   // reads of sK/sV[cur] done before next re-stage
    }

    // normalize by the WMMA-accumulated denominator and write attn f16
#pragma unroll
    for (int dc = 0; dc < 4; dc++)
#pragma unroll
        for (int vg = 0; vg < 8; vg++) {
            const int row = b * LL + f0 + vg + 8 * hf;
            const int col = head * DD + dc * 16 + lr;
            attnOut[(size_t)row * HH + col] = (f16)(O[dc][vg] / O5[vg]);
        }
}

// ---------------------------------------------------------------------------
extern "C" void kernel_launch(void* const* d_in, const int* in_sizes, int n_in,
                              void* d_out, int out_size, void* d_ws, size_t ws_size,
                              hipStream_t stream) {
    const float* xq   = (const float*)d_in[0];
    const float* xs   = (const float*)d_in[1];
    const float* bias = (const float*)d_in[2];
    const float* wq   = (const float*)d_in[3];
    const float* wk   = (const float*)d_in[4];
    const float* wv   = (const float*)d_in[5];
    const float* wo   = (const float*)d_in[6];

    unsigned short* ws = (unsigned short*)d_ws;
    const size_t XN = (size_t)MROWS * HH;   // 4,194,304
    const size_t WN = (size_t)HH * HH;      // 1,048,576
    unsigned short* xqh  = ws;  ws += XN;
    unsigned short* xsh  = ws;  ws += XN;
    unsigned short* wqT  = ws;  ws += WN;
    unsigned short* wkT  = ws;  ws += WN;
    unsigned short* wvT  = ws;  ws += WN;
    unsigned short* woT  = ws;  ws += WN;
    unsigned short* qh   = ws;  ws += XN;
    unsigned short* kh   = ws;  ws += XN;
    unsigned short* vt   = ws;  ws += XN;
    unsigned short* attn = ws;  ws += XN;

    // convert activations
    cvt_f16_kernel<<<(int)(XN / 256), 256, 0, stream>>>(xq, xqh, (int)XN);
    cvt_f16_kernel<<<(int)(XN / 256), 256, 0, stream>>>(xs, xsh, (int)XN);
    // transpose-convert weights so all GEMM B operands are [col][K]
    dim3 tg(32, 32);
    cvt_transpose_kernel<<<tg, 256, 0, stream>>>(wq, wqT);
    cvt_transpose_kernel<<<tg, 256, 0, stream>>>(wk, wkT);
    cvt_transpose_kernel<<<tg, 256, 0, stream>>>(wv, wvT);
    cvt_transpose_kernel<<<tg, 256, 0, stream>>>(wo, woT);

    dim3 gg(HH / 128, MROWS / 128);   // (8, 32)
    // Q scaled by log2e/sqrt(D) -> QK^T lands directly in the log2 domain
    gemm_f16_kernel<0><<<gg, 256, 0, stream>>>(xqh, wqT, qh, 0.125f * LOG2E);
    gemm_f16_kernel<0><<<gg, 256, 0, stream>>>(xsh, wkT, kh, 1.0f);
    gemm_f16_kernel<1><<<gg, 256, 0, stream>>>(xsh, wvT, vt, 1.0f);

    attn_kernel<<<dim3(LL / 128, NHEAD, BB), 256, 0, stream>>>(qh, kh, vt, bias, attn);

    // output projection -> f32 d_out
    gemm_f16_kernel<2><<<gg, 256, 0, stream>>>(attn, woT, d_out, 1.0f);
}
